// SimpleMoE_25598005084530
// MI455X (gfx1250) — matrix operand order repaired and checked by reading, not compiled, not run on hardware
//
#include <hip/hip_runtime.h>
#include <hip/hip_bf16.h>

// ---------------- problem constants (match reference) ----------------
static constexpr int DIM  = 1024;
static constexpr int FF   = 4096;
static constexpr int E    = 8;
static constexpr int TOK  = 2 * 2048;   // B*S
static constexpr int SLOTS = TOK * 2;   // top-2 routing
static constexpr int CAP  = TOK;        // max tokens one expert can receive

typedef __bf16 bf16x16 __attribute__((ext_vector_type(16)));
typedef float  f32x8   __attribute__((ext_vector_type(8)));

union Frag { uint4 q[2]; bf16x16 v; };

__device__ __forceinline__ unsigned short f2bf(float f) {
  unsigned int u = __float_as_uint(f);
  unsigned int r = (u + 0x7FFFu + ((u >> 16) & 1u)) >> 16;   // RNE
  return (unsigned short)r;
}

// ---------------- init: zero output + counters ----------------
__global__ void moe_init(float* out, int* counts, int* counts2) {
  int i = blockIdx.x * blockDim.x + threadIdx.x;
  if (i < TOK * DIM) out[i] = 0.0f;
  if (i < E) { counts[i] = 0; counts2[i] = 0; }
}

// ---------------- gate: logits, softmax, top-2 ----------------
__global__ void moe_gate(const float* __restrict__ x, const float* __restrict__ Wg,
                         const float* __restrict__ bg,
                         int* __restrict__ top_idx, float* __restrict__ top_w,
                         int* __restrict__ counts) {
  const int token = blockIdx.x;
  const int tid = threadIdx.x;
  const int wid = tid >> 5;       // expert handled by this wave (8 waves)
  const int lane = tid & 31;
  const float* xr = x + (size_t)token * DIM;
  float p = 0.0f;
  for (int d = lane; d < DIM; d += 32) p += xr[d] * Wg[d * E + wid];
  for (int off = 16; off > 0; off >>= 1) p += __shfl_xor(p, off, 32);
  __shared__ float sl[E];
  if (lane == 0) sl[wid] = p + bg[wid];
  __syncthreads();
  if (tid == 0) {
    float mx = sl[0];
    for (int e = 1; e < E; ++e) mx = fmaxf(mx, sl[e]);
    float pr[E]; float s = 0.0f;
    for (int e = 0; e < E; ++e) { pr[e] = __expf(sl[e] - mx); s += pr[e]; }
    float inv = 1.0f / s;
    float m1 = -1.0f, m2 = -1.0f; int i1 = 0, i2 = 0;
    for (int e = 0; e < E; ++e) {
      float v = pr[e] * inv;
      if (v > m1)      { m2 = m1; i2 = i1; m1 = v; i1 = e; }
      else if (v > m2) { m2 = v; i2 = e; }
    }
    float den = 1.0f / (m1 + m2 + 1e-9f);
    top_idx[token * 2 + 0] = i1;  top_w[token * 2 + 0] = m1 * den;
    top_idx[token * 2 + 1] = i2;  top_w[token * 2 + 1] = m2 * den;
    atomicAdd(&counts[i1], 1);
    atomicAdd(&counts[i2], 1);
  }
}

// ---------------- exclusive prefix over 8 counts ----------------
__global__ void moe_prefix(const int* __restrict__ counts, int* __restrict__ offs) {
  if (threadIdx.x == 0) {
    int a = 0;
    for (int e = 0; e < E; ++e) { offs[e] = a; a += counts[e]; }
    offs[E] = a;
  }
}

// ---------------- scatter tokens into compact per-expert slot lists ----------------
__global__ void moe_scatter(const int* __restrict__ top_idx, const float* __restrict__ top_w,
                            const int* __restrict__ offs, int* __restrict__ counts2,
                            int* __restrict__ slot_token, float* __restrict__ slot_w) {
  int t = blockIdx.x * blockDim.x + threadIdx.x;
  if (t >= TOK) return;
  for (int k = 0; k < 2; ++k) {
    int e = top_idx[t * 2 + k];
    int i = atomicAdd(&counts2[e], 1);
    int s = offs[e] + i;
    slot_token[s] = t;
    slot_w[s] = top_w[t * 2 + k];
  }
}

// ---------------- fp32 -> bf16 (x) ----------------
__global__ void moe_convert_x(const float* __restrict__ x, unsigned short* __restrict__ xb) {
  int i = blockIdx.x * blockDim.x + threadIdx.x;
  if (i < TOK * DIM) xb[i] = f2bf(x[i]);
}

// ---------------- fp32 [e][R][C] -> bf16 [e][C][R] (LDS-tiled transpose) ----------------
__global__ void moe_transpose_convert(const float* __restrict__ in, unsigned short* __restrict__ out,
                                      int R, int C) {
  __shared__ float tile[32][33];
  const int e = blockIdx.z;
  const int tx = threadIdx.x, ty = threadIdx.y;        // (32,8)
  const int c0 = blockIdx.x * 32, r0 = blockIdx.y * 32;
  const float* src = in + (size_t)e * R * C;
  unsigned short* dst = out + (size_t)e * R * C;
  for (int j = 0; j < 32; j += 8)
    tile[ty + j][tx] = src[(size_t)(r0 + ty + j) * C + (c0 + tx)];
  __syncthreads();
  for (int j = 0; j < 32; j += 8)
    dst[(size_t)(c0 + ty + j) * R + (r0 + tx)] = f2bf(tile[tx][ty + j]);
}

// ---------------- shared WMMA macro-tile compute (64x32 per wave) ----------------
// A/B LDS tiles are 128 rows x 32 K-elements (bf16), row stride 32 elements.
__device__ __forceinline__ void wmma_tile_compute(
    const unsigned short* __restrict__ As, const unsigned short* __restrict__ Bs,
    int waveM, int waveN, int lr, int lg, f32x8 acc[4][2]) {
  Frag a[4], b[2];
#pragma unroll
  for (int mf = 0; mf < 4; ++mf) {
    int row = waveM * 64 + mf * 16 + lr;
    int kb = lg * 8;
    a[mf].q[0] = *(const uint4*)&As[row * 32 + kb];
    a[mf].q[1] = *(const uint4*)&As[row * 32 + kb + 16];
  }
#pragma unroll
  for (int nf = 0; nf < 2; ++nf) {
    int n = waveN * 32 + nf * 16 + lr;
    int kb2 = lg * 8;
    b[nf].q[0] = *(const uint4*)&Bs[n * 32 + kb2];
    b[nf].q[1] = *(const uint4*)&Bs[n * 32 + kb2 + 16];
  }
#pragma unroll
  for (int mf = 0; mf < 4; ++mf)
#pragma unroll
    for (int nf = 0; nf < 2; ++nf)
      acc[mf][nf] = __builtin_amdgcn_wmma_f32_16x16x32_bf16(
          false, a[mf].v, false, b[nf].v, (short)0, acc[mf][nf], false, false);
}

// =====================================================================
// GEMM1: h[slot][FF] = relu( gather(xb)[Me x DIM] @ W1[e] + b1[e] ), bf16 out
// Tile: 128(M) x 128(N) x 32(K). 8 waves: 2(M) x 4(N), each 64x32 via WMMA.
// Double-buffered LDS, branch-free clamped gather (invalid rows masked in
// the epilogue; WMMA never traps), one barrier per K-step.
// =====================================================================
__global__ __launch_bounds__(256)
void moe_gemm1(const unsigned short* __restrict__ xb, const unsigned short* __restrict__ W1t,
               const float* __restrict__ b1, const int* __restrict__ counts,
               const int* __restrict__ offs, const int* __restrict__ slot_token,
               unsigned short* __restrict__ h) {
  const int e = blockIdx.z;
  const int count = counts[e];
  const int mbase = blockIdx.y * 128;
  if (mbase >= count) return;
  const int nbase = blockIdx.x * 128;
  const int sbase = offs[e];
  const int tid = threadIdx.x;

  __shared__ __align__(16) unsigned short As[2][128 * 32];
  __shared__ __align__(16) unsigned short Bs[2][128 * 32];
  __shared__ int rowTok[128];

  if (tid < 128) {
    int row = mbase + tid;
    rowTok[tid] = (row < count) ? slot_token[sbase + row] : 0;   // clamp, not mask
  }
  __syncthreads();

  // loop-invariant staging pointers: thread owns 2 16B chunks per tile
  const int idx0 = tid, idx1 = tid + 256;
  const int r0 = idx0 >> 2, c0 = (idx0 & 3) * 8;
  const int r1 = idx1 >> 2, c1 = (idx1 & 3) * 8;
  const unsigned short* pa0 = xb + (size_t)rowTok[r0] * DIM + c0;
  const unsigned short* pa1 = xb + (size_t)rowTok[r1] * DIM + c1;
  const unsigned short* pb0 = W1t + ((size_t)e * FF + nbase + r0) * DIM + c0;
  const unsigned short* pb1 = W1t + ((size_t)e * FF + nbase + r1) * DIM + c1;
  const int lds0 = r0 * 32 + c0, lds1 = r1 * 32 + c1;

  const int wid = tid >> 5, lane = tid & 31;
  const int waveM = wid >> 2, waveN = wid & 3;
  const int lr = lane & 15, lg = lane >> 4;

  f32x8 acc[4][2];
  const f32x8 zero = {0, 0, 0, 0, 0, 0, 0, 0};
  for (int m = 0; m < 4; ++m) for (int n = 0; n < 2; ++n) acc[m][n] = zero;

  // prologue: stage tile 0 into buffer 0
  *(uint4*)&As[0][lds0] = *(const uint4*)pa0;
  *(uint4*)&As[0][lds1] = *(const uint4*)pa1;
  *(uint4*)&Bs[0][lds0] = *(const uint4*)pb0;
  *(uint4*)&Bs[0][lds1] = *(const uint4*)pb1;
  pa0 += 32; pa1 += 32; pb0 += 32; pb1 += 32;
  __syncthreads();

  constexpr int KT = DIM / 32;
  for (int kt = 0; kt < KT; ++kt) {
    const int cur = kt & 1, nxt = cur ^ 1;
    uint4 va0, va1, vb0, vb1;
    const bool more = (kt + 1) < KT;
    if (more) {                         // issue next tile's global loads early
      va0 = *(const uint4*)pa0;  va1 = *(const uint4*)pa1;
      vb0 = *(const uint4*)pb0;  vb1 = *(const uint4*)pb1;
      pa0 += 32; pa1 += 32; pb0 += 32; pb1 += 32;
    }
    wmma_tile_compute(As[cur], Bs[cur], waveM, waveN, lr, lg, acc);
    if (more) {
      *(uint4*)&As[nxt][lds0] = va0;  *(uint4*)&As[nxt][lds1] = va1;
      *(uint4*)&Bs[nxt][lds0] = vb0;  *(uint4*)&Bs[nxt][lds1] = vb1;
    }
    __syncthreads();
  }

  // epilogue: bias + relu -> h (bf16)
  for (int mf = 0; mf < 4; ++mf)
    for (int nf = 0; nf < 2; ++nf)
      for (int j = 0; j < 8; ++j) {
        int M = waveM * 64 + mf * 16 + lg * 8 + j;
        int N = waveN * 32 + nf * 16 + lr;
        int row = mbase + M;
        if (row < count) {
          int col = nbase + N;
          float v = acc[mf][nf][j] + b1[e * FF + col];
          v = v > 0.0f ? v : 0.0f;
          h[(size_t)(sbase + row) * FF + col] = f2bf(v);
        }
      }
}

// =====================================================================
// GEMM2: out[token] += w_slot * ( h[slot][FF] @ W2[e] + b2[e] )
// Same tiling; A rows come straight from h (slot-contiguous, clamped).
// fp32 hardware atomics scatter into the zeroed output.
// =====================================================================
__global__ __launch_bounds__(256)
void moe_gemm2(const unsigned short* __restrict__ h, const unsigned short* __restrict__ W2t,
               const float* __restrict__ b2, const int* __restrict__ counts,
               const int* __restrict__ offs, const int* __restrict__ slot_token,
               const float* __restrict__ slot_w, float* __restrict__ out) {
  const int e = blockIdx.z;
  const int count = counts[e];
  const int mbase = blockIdx.y * 128;
  if (mbase >= count) return;
  const int nbase = blockIdx.x * 128;
  const int sbase = offs[e];
  const int tid = threadIdx.x;

  __shared__ __align__(16) unsigned short As[2][128 * 32];
  __shared__ __align__(16) unsigned short Bs[2][128 * 32];
  __shared__ int rowTok[128];
  __shared__ float rowW[128];

  if (tid < 128) {
    int row = mbase + tid;
    bool ok = row < count;
    rowTok[tid] = ok ? slot_token[sbase + row] : 0;
    rowW[tid]   = ok ? slot_w[sbase + row] : 0.0f;
  }
  __syncthreads();

  const int idx0 = tid, idx1 = tid + 256;
  const int r0 = idx0 >> 2, c0 = (idx0 & 3) * 8;
  const int r1 = idx1 >> 2, c1 = (idx1 & 3) * 8;
  const int s0 = min(sbase + mbase + r0, SLOTS - 1);   // clamp: garbage rows masked later
  const int s1 = min(sbase + mbase + r1, SLOTS - 1);
  const unsigned short* pa0 = h + (size_t)s0 * FF + c0;
  const unsigned short* pa1 = h + (size_t)s1 * FF + c1;
  const unsigned short* pb0 = W2t + ((size_t)e * DIM + nbase + r0) * FF + c0;
  const unsigned short* pb1 = W2t + ((size_t)e * DIM + nbase + r1) * FF + c1;
  const int lds0 = r0 * 32 + c0, lds1 = r1 * 32 + c1;

  const int wid = tid >> 5, lane = tid & 31;
  const int waveM = wid >> 2, waveN = wid & 3;
  const int lr = lane & 15, lg = lane >> 4;

  f32x8 acc[4][2];
  const f32x8 zero = {0, 0, 0, 0, 0, 0, 0, 0};
  for (int m = 0; m < 4; ++m) for (int n = 0; n < 2; ++n) acc[m][n] = zero;

  *(uint4*)&As[0][lds0] = *(const uint4*)pa0;
  *(uint4*)&As[0][lds1] = *(const uint4*)pa1;
  *(uint4*)&Bs[0][lds0] = *(const uint4*)pb0;
  *(uint4*)&Bs[0][lds1] = *(const uint4*)pb1;
  pa0 += 32; pa1 += 32; pb0 += 32; pb1 += 32;
  __syncthreads();

  constexpr int KT = FF / 32;
  for (int kt = 0; kt < KT; ++kt) {
    const int cur = kt & 1, nxt = cur ^ 1;
    uint4 va0, va1, vb0, vb1;
    const bool more = (kt + 1) < KT;
    if (more) {
      va0 = *(const uint4*)pa0;  va1 = *(const uint4*)pa1;
      vb0 = *(const uint4*)pb0;  vb1 = *(const uint4*)pb1;
      pa0 += 32; pa1 += 32; pb0 += 32; pb1 += 32;
    }
    wmma_tile_compute(As[cur], Bs[cur], waveM, waveN, lr, lg, acc);
    if (more) {
      *(uint4*)&As[nxt][lds0] = va0;  *(uint4*)&As[nxt][lds1] = va1;
      *(uint4*)&Bs[nxt][lds0] = vb0;  *(uint4*)&Bs[nxt][lds1] = vb1;
    }
    __syncthreads();
  }

  for (int mf = 0; mf < 4; ++mf)
    for (int nf = 0; nf < 2; ++nf)
      for (int j = 0; j < 8; ++j) {
        int M = waveM * 64 + mf * 16 + lg * 8 + j;
        int N = waveN * 32 + nf * 16 + lr;
        int row = mbase + M;
        if (row < count) {
          int col = nbase + N;
          float v = (acc[mf][nf][j] + b2[e * DIM + col]) * rowW[M];
          __hip_atomic_fetch_add(&out[(size_t)rowTok[M] * DIM + col], v,
                                 __ATOMIC_RELAXED, __HIP_MEMORY_SCOPE_AGENT);
        }
      }
}

// ---------------- host-side orchestration ----------------
extern "C" void kernel_launch(void* const* d_in, const int* in_sizes, int n_in,
                              void* d_out, int out_size, void* d_ws, size_t ws_size,
                              hipStream_t stream) {
  (void)in_sizes; (void)n_in; (void)out_size; (void)ws_size;
  const float* x  = (const float*)d_in[0];
  const float* Wg = (const float*)d_in[1];
  const float* bg = (const float*)d_in[2];
  const float* W1 = (const float*)d_in[3];
  const float* b1 = (const float*)d_in[4];
  const float* W2 = (const float*)d_in[5];
  const float* b2 = (const float*)d_in[6];
  float* out = (float*)d_out;

  char* ws = (char*)d_ws;
  auto al = [](size_t v) { return (v + 255) & ~(size_t)255; };
  size_t o = 0;
  int*   counts   = (int*)(ws + o);  o = al(o + E * 4);
  int*   counts2  = (int*)(ws + o);  o = al(o + E * 4);
  int*   offs     = (int*)(ws + o);  o = al(o + (E + 1) * 4);
  int*   top_idx  = (int*)(ws + o);  o = al(o + (size_t)TOK * 2 * 4);
  float* top_w    = (float*)(ws + o); o = al(o + (size_t)TOK * 2 * 4);
  int*   slot_tok = (int*)(ws + o);  o = al(o + (size_t)SLOTS * 4);
  float* slot_w   = (float*)(ws + o); o = al(o + (size_t)SLOTS * 4);
  unsigned short* xb  = (unsigned short*)(ws + o); o = al(o + (size_t)TOK * DIM * 2);
  unsigned short* W1t = (unsigned short*)(ws + o); o = al(o + (size_t)E * DIM * FF * 2);
  unsigned short* W2t = (unsigned short*)(ws + o); o = al(o + (size_t)E * DIM * FF * 2);
  unsigned short* hbuf = (unsigned short*)(ws + o); o = al(o + (size_t)SLOTS * FF * 2);

  // 1) zero output + counters
  moe_init<<<(TOK * DIM + 255) / 256, 256, 0, stream>>>(out, counts, counts2);
  // 2) gating (softmax + top-2) and expert counts
  moe_gate<<<TOK, 256, 0, stream>>>(x, Wg, bg, top_idx, top_w, counts);
  // 3) compact slot assignment
  moe_prefix<<<1, 32, 0, stream>>>(counts, offs);
  moe_scatter<<<(TOK + 255) / 256, 256, 0, stream>>>(top_idx, top_w, offs, counts2,
                                                     slot_tok, slot_w);
  // 4) precision/layout conversion (one bandwidth pass)
  moe_convert_x<<<(TOK * DIM + 255) / 256, 256, 0, stream>>>(x, xb);
  // W1: [e][DIM][FF] -> [e][FF][DIM]
  moe_transpose_convert<<<dim3(FF / 32, DIM / 32, E), dim3(32, 8), 0, stream>>>(W1, W1t, DIM, FF);
  // W2: [e][FF][DIM] -> [e][DIM][FF]
  moe_transpose_convert<<<dim3(DIM / 32, FF / 32, E), dim3(32, 8), 0, stream>>>(W2, W2t, FF, DIM);
  // 5) grouped GEMM1 (gather + bias + relu), bf16 WMMA
  moe_gemm1<<<dim3(FF / 128, CAP / 128, E), 256, 0, stream>>>(xb, W1t, b1, counts, offs,
                                                              slot_tok, hbuf);
  // 6) grouped GEMM2 (bias + routing weight + atomic scatter-add)
  moe_gemm2<<<dim3(DIM / 128, CAP / 128, E), 256, 0, stream>>>(hbuf, W2t, b2, counts, offs,
                                                               slot_tok, slot_w, out);
}
